// LSTMCellKFAC_45088566673863
// MI455X (gfx1250) — matrix operand order repaired and checked by reading, not compiled, not run on hardware
//
#include <hip/hip_runtime.h>
#include <stdint.h>

// ---- CDNA5 WMMA types ----
typedef __attribute__((ext_vector_type(16))) __bf16   v16bf;
typedef __attribute__((ext_vector_type(8)))  float    v8f;
typedef __attribute__((ext_vector_type(4)))  uint32_t u32x4;
typedef __attribute__((ext_vector_type(4)))  int      v4i;

#define BATCH 4096
#define HID   1024
#define INDIM 1024
#define KDIM  2048   // concat [input | h]
#define BM 128       // batch rows per block
#define BN 32        // hidden cols per block
#define KT 32        // K step in bf16 elements (WMMA K)
#define KU 16        // K step in uint32 (packed 2x bf16)
#define LDA 20       // uint32 per LDS A row = 40 bf16 (32 data + 8 pad)
#define LDB 20       // uint32 per LDS B row
#define KSTEPS (KDIM / KT)   // 64

// Async global->LDS copy path (gfx1250): guard so either toolchain compiles.
// Builtin signature (from clang diagnostic): (int4 AS1*, int4 AS3*, imm, imm)
#if defined(__gfx1250__) && __has_builtin(__builtin_amdgcn_global_load_async_to_lds_b128) && __has_builtin(__builtin_amdgcn_s_wait_asynccnt)
#define USE_ASYNC_LDS 1
#define AS_GLOBAL(p) ((__attribute__((address_space(1))) v4i*)(p))
#define AS_LDS(p)    ((__attribute__((address_space(3))) v4i*)(p))
#else
#define USE_ASYNC_LDS 0
#endif

// fp32 -> packed 2x bf16, round-to-nearest-even
__device__ __forceinline__ uint32_t pk_bf16(float lo, float hi) {
  uint32_t a = __float_as_uint(lo);
  uint32_t b = __float_as_uint(hi);
  a = (a + 0x7FFFu + ((a >> 16) & 1u)) >> 16;
  b = (b + 0x7FFFu + ((b >> 16) & 1u)) >> 16;
  return (a & 0xFFFFu) | (b << 16);
}

__device__ __forceinline__ float fast_sigmoid(float x) {
  x = fminf(fmaxf(x, -30.f), 30.f);
  return 1.f / (1.f + __expf(-x));
}
__device__ __forceinline__ float fast_tanh(float x) {
  x = fminf(fmaxf(x, -15.f), 15.f);
  float e = __expf(2.f * x);
  return (e - 1.f) / (e + 1.f);
}

// ---------------------------------------------------------------------------
// Pre-pass: convert an (rows x cols) fp32 matrix into packed-bf16 u32 words at
// dst[row * dst_stride_u32 + dst_off_u32 + ...]. One float4 per thread.
// ---------------------------------------------------------------------------
__global__ __launch_bounds__(256)
void cvt_f32_to_bf16(const float* __restrict__ src, uint32_t* __restrict__ dst,
                     int rows, int cols_v4, int dst_stride_u32, int dst_off_u32)
{
  int idx = blockIdx.x * 256 + threadIdx.x;
  if (idx >= rows * cols_v4) return;
  int r  = idx / cols_v4;
  int cv = idx - r * cols_v4;
  float4 v = reinterpret_cast<const float4*>(src)[(size_t)r * cols_v4 + cv];
  uint2 p;
  p.x = pk_bf16(v.x, v.y);
  p.y = pk_bf16(v.z, v.w);
  *reinterpret_cast<uint2*>(dst + (size_t)r * dst_stride_u32 + dst_off_u32 + 2 * cv) = p;
}

// ---------------------------------------------------------------------------
// Fused 4-gate LSTM GEMM: gates = [x|h] @ [Wi|Wh]^T, epilogue in-register.
//   Abuf: 4096 rows x 1024 u32 (2048 bf16: K = [input | h])
//   Wbuf: (4 gates * 1024) rows x 1024 u32 (2048 bf16: K = [w_i | w_h])
// ---------------------------------------------------------------------------
__global__ __launch_bounds__(256)
void lstm_gemm_wmma(const uint32_t* __restrict__ Abuf,
                    const uint32_t* __restrict__ Wbuf,
                    const float* __restrict__ c,
                    const float* __restrict__ bii, const float* __restrict__ bih,
                    const float* __restrict__ bfi, const float* __restrict__ bfh,
                    const float* __restrict__ bgi, const float* __restrict__ bgh,
                    const float* __restrict__ boi, const float* __restrict__ boh,
                    float* __restrict__ hp_out, float* __restrict__ cp_out)
{
  __shared__ uint32_t shA[2][BM * LDA];       // 2 x 10 KB
  __shared__ uint32_t shB[2][4 * BN * LDB];   // 2 x 10 KB

  const int tid  = threadIdx.x;
  const int lane = tid & 31;
  const int wave = tid >> 5;       // 8 waves
  const int hi   = lane >> 4;      // half-wave select (CDNA5 wave32 frag layout)
  const int ln16 = lane & 15;

  const int rowBase = blockIdx.x * BM;
  const int colBase = blockIdx.y * BN;

  // staging maps: each thread moves 8 u32 (= 2 x b128) for A and for B
  const int a_row   = tid >> 1;          // 0..127
  const int a_chunk = tid & 1;
  const int b_gate  = tid >> 6;          // 0..3 (uniform per wave)
  const int b_n     = (tid & 63) >> 1;   // 0..31
  const int b_chunk = tid & 1;

  const uint32_t* gA = Abuf + (size_t)(rowBase + a_row) * 1024 + a_chunk * 8;
  const uint32_t* gB = Wbuf + (size_t)((b_gate << 10) + colBase + b_n) * 1024 + b_chunk * 8;

  // wave tiling: 4 waves along M (32 rows -> 2 frags), 2 waves along N (16 cols)
  const int wy = wave & 3;
  const int wx = wave >> 2;

  v8f acc_i0 = {}; v8f acc_i1 = {};
  v8f acc_f0 = {}; v8f acc_f1 = {};
  v8f acc_g0 = {}; v8f acc_g1 = {};
  v8f acc_o0 = {}; v8f acc_o1 = {};

  union Frag { u32x4 q[2]; v16bf v; };

  auto stage = [&](int kt, int buf) {
    const uint32_t* ga = gA + kt * KU;
    const uint32_t* gb = gB + kt * KU;
    uint32_t* la = &shA[buf][a_row * LDA + a_chunk * 8];
    uint32_t* lb = &shB[buf][(b_gate * BN + b_n) * LDB + b_chunk * 8];
#if USE_ASYNC_LDS
    __builtin_amdgcn_global_load_async_to_lds_b128(AS_GLOBAL(ga), AS_LDS(la), 0, 0);
    __builtin_amdgcn_global_load_async_to_lds_b128(AS_GLOBAL(ga), AS_LDS(la), 16, 0);
    __builtin_amdgcn_global_load_async_to_lds_b128(AS_GLOBAL(gb), AS_LDS(lb), 0, 0);
    __builtin_amdgcn_global_load_async_to_lds_b128(AS_GLOBAL(gb), AS_LDS(lb), 16, 0);
#else
    u32x4 ta0 = *reinterpret_cast<const u32x4*>(ga);
    u32x4 ta1 = *reinterpret_cast<const u32x4*>(ga + 4);
    u32x4 tb0 = *reinterpret_cast<const u32x4*>(gb);
    u32x4 tb1 = *reinterpret_cast<const u32x4*>(gb + 4);
    *reinterpret_cast<u32x4*>(la)     = ta0;
    *reinterpret_cast<u32x4*>(la + 4) = ta1;
    *reinterpret_cast<u32x4*>(lb)     = tb0;
    *reinterpret_cast<u32x4*>(lb + 4) = tb1;
#endif
  };

  stage(0, 0);   // prologue: fill buffer 0

  for (int kt = 0; kt < KSTEPS; ++kt) {
    const int cur = kt & 1;
#if USE_ASYNC_LDS
    __builtin_amdgcn_s_wait_asynccnt(0);   // my async copies for stage kt landed
#endif
    __syncthreads();                       // everyone's landed; prev buf reads done

    if (kt + 1 < KSTEPS) stage(kt + 1, cur ^ 1);   // overlap DMA with WMMAs below

    // ---- A fragments (16x32 bf16; lane = M, 16B chunks at K 8*hi, 16+8*hi) ----
    Frag a0, a1, b;
    {
      const uint32_t* p0 = &shA[cur][(wy * 32 + ln16) * LDA + hi * 4];
      a0.q[0] = *reinterpret_cast<const u32x4*>(p0);
      a0.q[1] = *reinterpret_cast<const u32x4*>(p0 + 8);
      const uint32_t* p1 = &shA[cur][(wy * 32 + 16 + ln16) * LDA + hi * 4];
      a1.q[0] = *reinterpret_cast<const u32x4*>(p1);
      a1.q[1] = *reinterpret_cast<const u32x4*>(p1 + 8);
    }
    const int nloc = wx * 16 + ln16;   // B: lane = N, 16B chunks at K 8*hi, 16+8*hi

    #define GATE_MMA(G, A0, A1) do {                                          \
      const uint32_t* pb = &shB[cur][((G) * BN + nloc) * LDB + hi * 4];       \
      b.q[0] = *reinterpret_cast<const u32x4*>(pb);                           \
      b.q[1] = *reinterpret_cast<const u32x4*>(pb + 8);                       \
      A0 = __builtin_amdgcn_wmma_f32_16x16x32_bf16(false, a0.v, false, b.v,   \
                                                   (short)0, A0, false, false);\
      A1 = __builtin_amdgcn_wmma_f32_16x16x32_bf16(false, a1.v, false, b.v,   \
                                                   (short)0, A1, false, false);\
    } while (0)

    GATE_MMA(0, acc_i0, acc_i1);
    GATE_MMA(1, acc_f0, acc_f1);
    GATE_MMA(2, acc_g0, acc_g1);
    GATE_MMA(3, acc_o0, acc_o1);
    #undef GATE_MMA
  }

  // ---- epilogue: fused biases + LSTM gating in-register ----
  // C/D layout: element e of v8f at (M = e + 8*hi, N = lane&15)
  const int col = colBase + wx * 16 + ln16;
  const float bi = bii[col] + bih[col];
  const float bf = bfi[col] + bfh[col];
  const float bg = bgi[col] + bgh[col];
  const float bo = boi[col] + boh[col];

  #pragma unroll
  for (int mi = 0; mi < 2; ++mi) {
    v8f ai = mi ? acc_i1 : acc_i0;
    v8f af = mi ? acc_f1 : acc_f0;
    v8f ag = mi ? acc_g1 : acc_g0;
    v8f ao = mi ? acc_o1 : acc_o0;
    #pragma unroll
    for (int e = 0; e < 8; ++e) {
      const int row = rowBase + wy * 32 + mi * 16 + e + 8 * hi;
      const size_t idx = (size_t)row * HID + col;
      float iv = fast_sigmoid(ai[e] + bi);
      float fv = fast_sigmoid(af[e] + bf);
      float gv = fast_tanh  (ag[e] + bg);
      float ov = fast_sigmoid(ao[e] + bo);
      float cpv = fv * c[idx] + iv * gv;
      float hpv = ov * fast_tanh(cpv);
      hp_out[idx] = hpv;
      cp_out[idx] = cpv;
    }
  }
}

extern "C" void kernel_launch(void* const* d_in, const int* in_sizes, int n_in,
                              void* d_out, int out_size, void* d_ws, size_t ws_size,
                              hipStream_t stream) {
  (void)in_sizes; (void)n_in; (void)out_size; (void)ws_size;
  const float* input = (const float*)d_in[0];
  const float* h     = (const float*)d_in[1];
  const float* c     = (const float*)d_in[2];
  const float* wii = (const float*)d_in[3];  const float* bii = (const float*)d_in[4];
  const float* wih = (const float*)d_in[5];  const float* bih = (const float*)d_in[6];
  const float* wfi = (const float*)d_in[7];  const float* bfi = (const float*)d_in[8];
  const float* wfh = (const float*)d_in[9];  const float* bfh = (const float*)d_in[10];
  const float* wgi = (const float*)d_in[11]; const float* bgi = (const float*)d_in[12];
  const float* wgh = (const float*)d_in[13]; const float* bgh = (const float*)d_in[14];
  const float* woi = (const float*)d_in[15]; const float* boi = (const float*)d_in[16];
  const float* woh = (const float*)d_in[17]; const float* boh = (const float*)d_in[18];

  // Workspace layout (packed bf16 as u32 words):
  //   Abuf: 4096 rows x 1024 u32  (K = [input | h])           16 MB
  //   Wbuf: 4096 rows x 1024 u32  (4 gates x [w_i | w_h])     16 MB
  uint32_t* Abuf = (uint32_t*)d_ws;
  uint32_t* Wbuf = Abuf + (size_t)BATCH * (KDIM / 2);

  auto cvt = [&](const float* s, uint32_t* dbase, int rows, int off_u32) {
    int total = rows * (1024 / 4);
    cvt_f32_to_bf16<<<(total + 255) / 256, dim3(256), 0, stream>>>(
        s, dbase, rows, 1024 / 4, KDIM / 2, off_u32);
  };
  cvt(input, Abuf, BATCH, 0);
  cvt(h,     Abuf, BATCH, 512);
  const float* wI[4] = { wii, wfi, wgi, woi };
  const float* wH[4] = { wih, wfh, wgh, woh };
  for (int g = 0; g < 4; ++g) {
    uint32_t* wdst = Wbuf + (size_t)g * HID * (KDIM / 2);
    cvt(wI[g], wdst, HID, 0);
    cvt(wH[g], wdst, HID, 512);
  }

  float* hp = (float*)d_out;                 // h' first (tuple order)
  float* cp = hp + (size_t)BATCH * HID;      // then c'

  dim3 grid(BATCH / BM, HID / BN);           // 32 x 32 = 1024 blocks
  lstm_gemm_wmma<<<grid, dim3(256), 0, stream>>>(
      Abuf, Wbuf, c,
      bii, bih, bfi, bfh, bgi, bgh, boi, boh,
      hp, cp);
}
